// TorchLMHeadJSD_85607288144570
// MI455X (gfx1250) — matrix operand order repaired and checked by reading, not compiled, not run on hardware
//
#include <hip/hip_runtime.h>
#include <hip/hip_bf16.h>
#include <math.h>

// Problem constants (from the reference)
#define H_T 2048          // teacher hidden
#define H_S 1024          // student hidden
#define VOCAB 32000
#define NTOK 2048
#define IGNORE_INDEX (-100)

// Tiling
#define MBLK 32                       // token rows per workgroup (2 M-tiles)
#define NROWBLK (NTOK / MBLK)         // 64
#define NCHUNK 20                     // vocab chunks
#define CHUNK_COLS (VOCAB / NCHUNK)   // 1600
#define NPAIRS (CHUNK_COLS / 32)      // 50 pairs of 16-col tiles
#define THREADS 256
#define NWAVES (THREADS / 32)

typedef __attribute__((ext_vector_type(16))) __bf16 v16bf;
typedef __attribute__((ext_vector_type(8)))  float  v8f;

union Frag16 { uint4 u[2]; v16bf v; };

__device__ __forceinline__ unsigned int pack2_bf16(float a, float b) {
  union { float f; unsigned int u; } xa, xb;
  xa.f = a; xb.f = b;
  unsigned int ra = (xa.u + 0x7FFFu + ((xa.u >> 16) & 1u)) >> 16;
  unsigned int rb = (xb.u + 0x7FFFu + ((xb.u >> 16) & 1u)) >> 16;
  return ra | (rb << 16);
}

__device__ __forceinline__ void olsm_merge(float& m, float& s, float om, float os) {
  float nm = fmaxf(m, om);
  s = s * __expf(m - nm) + os * __expf(om - nm);
  m = nm;
}

// B fragment: 16 consecutive bf16 per lane (32B)
__device__ __forceinline__ void load_g(Frag16& f, const unsigned short* p) {
  f.u[0] = *(const uint4*)(p);
  f.u[1] = *(const uint4*)(p + 8);
}
// A fragment: 16 consecutive bf16 per lane (32B)
__device__ __forceinline__ void load_a(Frag16& f, const unsigned short* p) {
  f.u[0] = *(const uint4*)(p);
  f.u[1] = *(const uint4*)(p + 8);
}

__device__ __forceinline__ v8f wmma_bf16(const Frag16& a, const Frag16& b, v8f c) {
  return __builtin_amdgcn_wmma_f32_16x16x32_bf16(false, a.v, false, b.v,
                                                 (short)0, c, false, false);
}

// ---------------------------------------------------------------------------
// Kernel 1: convert weights fp32 -> bf16 (vectorized); zero accumulators
// ---------------------------------------------------------------------------
__global__ void k_prepare(const float* __restrict__ sW, const float* __restrict__ tW,
                          unsigned short* __restrict__ sWb, unsigned short* __restrict__ tWb,
                          float* __restrict__ tgt_logit, float* __restrict__ accums) {
  size_t i = (size_t)blockIdx.x * blockDim.x + threadIdx.x;
  size_t stride = (size_t)gridDim.x * blockDim.x;
  const size_t nS4 = (size_t)VOCAB * H_S / 4;
  const size_t nT4 = (size_t)VOCAB * H_T / 4;
  for (size_t j = i; j < nS4; j += stride) {
    float4 f = ((const float4*)sW)[j];
    uint2 p; p.x = pack2_bf16(f.x, f.y); p.y = pack2_bf16(f.z, f.w);
    ((uint2*)sWb)[j] = p;
  }
  for (size_t j = i; j < nT4; j += stride) {
    float4 f = ((const float4*)tW)[j];
    uint2 p; p.x = pack2_bf16(f.x, f.y); p.y = pack2_bf16(f.z, f.w);
    ((uint2*)tWb)[j] = p;
  }
  for (size_t j = i; j < NTOK; j += stride) tgt_logit[j] = 0.f;
  if (i < 4) accums[i] = 0.f;
}

// ---------------------------------------------------------------------------
// Main WMMA sweep. PASS==1: online-softmax stats per (row, chunk) + target
// logit capture. PASS==2: recompute logits, accumulate JSD terms.
// Grid: (NROWBLK, NCHUNK), block: 256 threads (8 waves).
// Each wave iteration computes a 2Mx2N tile quad (32 rows x 32 cols): every
// A and B fragment feeds two WMMAs (16 FLOP/byte of LDS/L2 traffic), with
// one-deep software pipelining of the fragment loads.
// ---------------------------------------------------------------------------
template <int PASS>
__global__ __launch_bounds__(THREADS)
void k_gemm(const float* __restrict__ sIn, const float* __restrict__ tIn,
            const unsigned short* __restrict__ sWb, const unsigned short* __restrict__ tWb,
            const int* __restrict__ target,
            float4* __restrict__ stats,          // [NTOK][NCHUNK]
            float* __restrict__ tgt_logit,       // [NTOK]
            const float* __restrict__ lse_s, const float* __restrict__ lse_t,
            float* __restrict__ accums) {
  extern __shared__ char smem_raw[];
  unsigned short* As = (unsigned short*)smem_raw;        // MBLK x H_S bf16
  unsigned short* At = As + MBLK * H_S;                  // MBLK x H_T bf16
  float* red = (float*)(At + MBLK * H_T);                // [NWAVES][MBLK][4]
  float* lrow = red + NWAVES * MBLK * 4;                 // MBLK lse_s + MBLK lse_t
  int*   ttgt = (int*)(lrow + 2 * MBLK);                 // MBLK targets

  const int tid  = threadIdx.x;
  const int lane = tid & 31;
  const int wave = tid >> 5;
  const int ln   = lane & 15;           // B column / A row within tile
  const int hi   = (lane >> 4) & 1;     // lane half selector
  const int rowblk = blockIdx.x;        // fast dim: blocks sharing a chunk run together (L2)
  const int chunk  = blockIdx.y;
  const int row0 = rowblk * MBLK;
  const int col0 = chunk * CHUNK_COLS;

  // Stage A (inputs) into LDS, converting fp32 -> bf16 (vectorized x4).
  for (int idx = tid; idx < MBLK * H_S / 4; idx += THREADS) {
    int r = idx >> 8, kq = idx & (H_S / 4 - 1);
    float4 f = *(const float4*)(sIn + (size_t)(row0 + r) * H_S + kq * 4);
    uint2 p; p.x = pack2_bf16(f.x, f.y); p.y = pack2_bf16(f.z, f.w);
    *(uint2*)(As + idx * 4) = p;
  }
  for (int idx = tid; idx < MBLK * H_T / 4; idx += THREADS) {
    int r = idx >> 9, kq = idx & (H_T / 4 - 1);
    float4 f = *(const float4*)(tIn + (size_t)(row0 + r) * H_T + kq * 4);
    uint2 p; p.x = pack2_bf16(f.x, f.y); p.y = pack2_bf16(f.z, f.w);
    *(uint2*)(At + idx * 4) = p;
  }
  if (tid < MBLK) {
    ttgt[tid] = target[row0 + tid];
    if (PASS == 2) {
      lrow[tid]        = lse_s[row0 + tid];
      lrow[MBLK + tid] = lse_t[row0 + tid];
    }
  }
  __syncthreads();

  // Per-lane online-softmax state: mi-th M-tile, VGPR r holds row mi*16 + r + 8*hi.
  float sm[2][8], ss[2][8], tm[2][8], ts[2][8];
  if (PASS == 1) {
    for (int mi = 0; mi < 2; ++mi)
      for (int r = 0; r < 8; ++r) {
        sm[mi][r] = -__builtin_inff(); ss[mi][r] = 0.f;
        tm[mi][r] = -__builtin_inff(); ts[mi][r] = 0.f;
      }
  }
  float kl_acc = 0.f;

  const unsigned short* AbS0 = As + ln * H_S + 16 * hi;         // rows 0..15
  const unsigned short* AbS1 = AbS0 + (size_t)16 * H_S;         // rows 16..31
  const unsigned short* AbT0 = At + ln * H_T + 16 * hi;
  const unsigned short* AbT1 = AbT0 + (size_t)16 * H_T;

  for (int jp = wave; jp < NPAIRS; jp += NWAVES) {
    const int n0 = col0 + jp * 32;

    // Prefetch next pair's weight rows toward L2 while this pair computes.
    {
      int jn = jp + NWAVES;
      if (jn < NPAIRS) {
        __builtin_prefetch(sWb + (size_t)(col0 + jn * 32 + ln) * H_S, 0, 1);
        __builtin_prefetch(tWb + (size_t)(col0 + jn * 32 + ln) * H_T, 0, 1);
      }
    }

    v8f aS[2][2] = {{{}, {}}, {{}, {}}};   // [mi][nj]
    v8f aT[2][2] = {{{}, {}}, {{}, {}}};

    // ---- student GEMM: K = 1024, 2Mx2N quad, 1-deep pipelined ----
    {
      const unsigned short* B0 = sWb + (size_t)(n0 + ln) * H_S + 16 * hi;
      const unsigned short* B1 = B0 + (size_t)16 * H_S;
      Frag16 a0, a1, b0, b1, a0n, a1n, b0n, b1n;
      load_a(a0, AbS0); load_a(a1, AbS1); load_g(b0, B0); load_g(b1, B1);
      for (int k0 = 32; k0 < H_S; k0 += 32) {
        load_a(a0n, AbS0 + k0); load_a(a1n, AbS1 + k0);
        load_g(b0n, B0 + k0);   load_g(b1n, B1 + k0);
        aS[0][0] = wmma_bf16(a0, b0, aS[0][0]);
        aS[0][1] = wmma_bf16(a0, b1, aS[0][1]);
        aS[1][0] = wmma_bf16(a1, b0, aS[1][0]);
        aS[1][1] = wmma_bf16(a1, b1, aS[1][1]);
        a0 = a0n; a1 = a1n; b0 = b0n; b1 = b1n;
      }
      aS[0][0] = wmma_bf16(a0, b0, aS[0][0]);
      aS[0][1] = wmma_bf16(a0, b1, aS[0][1]);
      aS[1][0] = wmma_bf16(a1, b0, aS[1][0]);
      aS[1][1] = wmma_bf16(a1, b1, aS[1][1]);
    }
    // ---- teacher GEMM: K = 2048, 2Mx2N quad, 1-deep pipelined ----
    {
      const unsigned short* B0 = tWb + (size_t)(n0 + ln) * H_T + 16 * hi;
      const unsigned short* B1 = B0 + (size_t)16 * H_T;
      Frag16 a0, a1, b0, b1, a0n, a1n, b0n, b1n;
      load_a(a0, AbT0); load_a(a1, AbT1); load_g(b0, B0); load_g(b1, B1);
      for (int k0 = 32; k0 < H_T; k0 += 32) {
        load_a(a0n, AbT0 + k0); load_a(a1n, AbT1 + k0);
        load_g(b0n, B0 + k0);   load_g(b1n, B1 + k0);
        aT[0][0] = wmma_bf16(a0, b0, aT[0][0]);
        aT[0][1] = wmma_bf16(a0, b1, aT[0][1]);
        aT[1][0] = wmma_bf16(a1, b0, aT[1][0]);
        aT[1][1] = wmma_bf16(a1, b1, aT[1][1]);
        a0 = a0n; a1 = a1n; b0 = b0n; b1 = b1n;
      }
      aT[0][0] = wmma_bf16(a0, b0, aT[0][0]);
      aT[0][1] = wmma_bf16(a0, b1, aT[0][1]);
      aT[1][0] = wmma_bf16(a1, b0, aT[1][0]);
      aT[1][1] = wmma_bf16(a1, b1, aT[1][1]);
    }

    if (PASS == 1) {
      for (int mi = 0; mi < 2; ++mi) {
        for (int r = 0; r < 8; ++r) {
          olsm_merge(sm[mi][r], ss[mi][r], aS[mi][0][r], 1.f);
          olsm_merge(sm[mi][r], ss[mi][r], aS[mi][1][r], 1.f);
          olsm_merge(tm[mi][r], ts[mi][r], aT[mi][0][r], 1.f);
          olsm_merge(tm[mi][r], ts[mi][r], aT[mi][1][r], 1.f);
          int row_local = mi * 16 + r + 8 * hi;
          int tg = ttgt[row_local];
          if (n0 + ln == tg)      tgt_logit[row0 + row_local] = aS[mi][0][r]; // unique writer
          if (n0 + 16 + ln == tg) tgt_logit[row0 + row_local] = aS[mi][1][r]; // unique writer
        }
      }
    } else {
      for (int mi = 0; mi < 2; ++mi) {
        for (int r = 0; r < 8; ++r) {
          int row_local = mi * 16 + r + 8 * hi;
          float lsS = lrow[row_local], lsT = lrow[MBLK + row_local];
          for (int nj = 0; nj < 2; ++nj) {
            float lps = aS[mi][nj][r] - lsS, lpt = aT[mi][nj][r] - lsT;
            float ps = __expf(lps), pt = __expf(lpt);
            float m = 0.5f * (ps + pt);
            float logm = __logf(fmaxf(m, 1e-38f));
            kl_acc += ps * (lps - logm) + pt * (lpt - logm);
          }
        }
      }
    }
  }

  if (PASS == 1) {
    // Merge stats across the 16 lanes sharing each row (xor over low 4 bits keeps M fixed).
    for (int msk = 1; msk < 16; msk <<= 1) {
      for (int mi = 0; mi < 2; ++mi) {
        for (int r = 0; r < 8; ++r) {
          float om = __shfl_xor(sm[mi][r], msk, 32);
          float os = __shfl_xor(ss[mi][r], msk, 32);
          olsm_merge(sm[mi][r], ss[mi][r], om, os);
          om = __shfl_xor(tm[mi][r], msk, 32);
          os = __shfl_xor(ts[mi][r], msk, 32);
          olsm_merge(tm[mi][r], ts[mi][r], om, os);
        }
      }
    }
    if (ln == 0) {
      for (int mi = 0; mi < 2; ++mi) {
        for (int r = 0; r < 8; ++r) {
          int row_local = mi * 16 + r + 8 * hi;
          float* p = red + ((wave * MBLK) + row_local) * 4;
          p[0] = sm[mi][r]; p[1] = ss[mi][r]; p[2] = tm[mi][r]; p[3] = ts[mi][r];
        }
      }
    }
    __syncthreads();
    if (tid < MBLK) {                 // one thread per row merges the 8 wave slots
      float bsm = -__builtin_inff(), bss = 0.f, btm = -__builtin_inff(), bts = 0.f;
      for (int w = 0; w < NWAVES; ++w) {
        const float* p = red + ((w * MBLK) + tid) * 4;
        olsm_merge(bsm, bss, p[0], p[1]);
        olsm_merge(btm, bts, p[2], p[3]);
      }
      stats[(size_t)(row0 + tid) * NCHUNK + chunk] = make_float4(bsm, bss, btm, bts);
    }
  } else {
    for (int msk = 1; msk < 32; msk <<= 1) kl_acc += __shfl_xor(kl_acc, msk, 32);
    if (lane == 0) red[wave] = kl_acc;
    __syncthreads();
    if (tid == 0) {
      float s = 0.f;
      for (int w = 0; w < NWAVES; ++w) s += red[w];
      atomicAdd(&accums[0], s);
    }
  }
}

// ---------------------------------------------------------------------------
// Kernel 3: reduce per-chunk stats -> per-row logsumexp; hard-loss accumulation
// ---------------------------------------------------------------------------
__global__ void k_rowreduce(const float4* __restrict__ stats,
                            const float* __restrict__ tgt_logit,
                            const int* __restrict__ target,
                            float* __restrict__ lse_s, float* __restrict__ lse_t,
                            float* __restrict__ accums) {
  int row = blockIdx.x * blockDim.x + threadIdx.x;
  if (row >= NTOK) return;
  float sm = -__builtin_inff(), ss = 0.f, tm = -__builtin_inff(), ts = 0.f;
  for (int c = 0; c < NCHUNK; ++c) {
    float4 p = stats[(size_t)row * NCHUNK + c];
    olsm_merge(sm, ss, p.x, p.y);
    olsm_merge(tm, ts, p.z, p.w);
  }
  float ls = sm + __logf(ss);
  float lt = tm + __logf(ts);
  lse_s[row] = ls;
  lse_t[row] = lt;
  int tg = target[row];
  bool valid = (tg != IGNORE_INDEX);
  atomicAdd(&accums[1], valid ? (ls - tgt_logit[row]) : 0.f);
  atomicAdd(&accums[2], valid ? 1.f : 0.f);
}

// ---------------------------------------------------------------------------
// Kernel 5: scalar combine
// ---------------------------------------------------------------------------
__global__ void k_finalize(const float* __restrict__ accums, float* __restrict__ out) {
  float kl = accums[0];
  float hard = accums[1] / fmaxf(accums[2], 1.f);
  float jsd = 0.5f * kl / (float)NTOK;   // (student_kl + teacher_kl)/2, each /N
  out[0] = 0.5f * hard + 0.5f * jsd;     // W_HARD * hard + W_SOFT * jsd
}

// ---------------------------------------------------------------------------
extern "C" void kernel_launch(void* const* d_in, const int* in_sizes, int n_in,
                              void* d_out, int out_size, void* d_ws, size_t ws_size,
                              hipStream_t stream) {
  const float* sIn = (const float*)d_in[0];   // (N, 1024)
  const float* tIn = (const float*)d_in[1];   // (N, 2048)
  const float* sW  = (const float*)d_in[2];   // (V, 1024)
  const float* tW  = (const float*)d_in[3];   // (V, 2048)
  const int*   tgt = (const int*)d_in[4];     // (N,)
  float* out = (float*)d_out;

  // Workspace carve (all 16B aligned)
  char* ws = (char*)d_ws;
  size_t off = 0;
  unsigned short* sWb = (unsigned short*)(ws + off); off += (size_t)VOCAB * H_S * 2;
  unsigned short* tWb = (unsigned short*)(ws + off); off += (size_t)VOCAB * H_T * 2;
  float4* stats   = (float4*)(ws + off); off += (size_t)NTOK * NCHUNK * sizeof(float4);
  float* lse_s    = (float*)(ws + off);  off += (size_t)NTOK * sizeof(float);
  float* lse_t    = (float*)(ws + off);  off += (size_t)NTOK * sizeof(float);
  float* tgt_log  = (float*)(ws + off);  off += (size_t)NTOK * sizeof(float);
  float* accums   = (float*)(ws + off);  off += 4 * sizeof(float);

  const size_t smem = (size_t)MBLK * H_S * 2 + (size_t)MBLK * H_T * 2 +
                      (size_t)NWAVES * MBLK * 4 * sizeof(float) +
                      2 * MBLK * sizeof(float) + MBLK * sizeof(int);

  k_prepare<<<4096, 256, 0, stream>>>(sW, tW, sWb, tWb, tgt_log, accums);

  dim3 grid(NROWBLK, NCHUNK);
  k_gemm<1><<<grid, THREADS, smem, stream>>>(sIn, tIn, sWb, tWb, tgt,
                                             stats, tgt_log, lse_s, lse_t, accums);

  k_rowreduce<<<(NTOK + 255) / 256, 256, 0, stream>>>(stats, tgt_log, tgt,
                                                      lse_s, lse_t, accums);

  k_gemm<2><<<grid, THREADS, smem, stream>>>(sIn, tIn, sWb, tWb, tgt,
                                             stats, tgt_log, lse_s, lse_t, accums);

  k_finalize<<<1, 1, 0, stream>>>(accums, out);
}
